// SpatialEncoder_24996709663003
// MI455X (gfx1250) — matrix operand order repaired and checked by reading, not compiled, not run on hardware
//
#include <hip/hip_runtime.h>
#include <hip/hip_bf16.h>

// ---------------- problem constants (from reference) ----------------
#define N_NODES   10000
#define MPAD      10240           // N_NODES padded to multiple of BM (256)
#define N_EDGES   160000
#define E_TOT     170000          // edges + self loops
#define IN_CH     4097
#define KPAD1     4128            // IN_CH padded to multiple of 32
#define HID       512
#define HEADS     4
#define F1        (HEADS * HID)   // 2048
#define OUT_CH    256
#define N_GRAPHS  64
#define NEG_SLOPE 0.2f

// ---------------- WMMA types ----------------
typedef __attribute__((ext_vector_type(16))) __bf16        v16bf;
typedef __attribute__((ext_vector_type(8)))  float         v8f;
typedef __attribute__((ext_vector_type(8)))  unsigned int  v8u;

union Frag { v8u u; v16bf b; };

// ---------------- helpers ----------------
__device__ inline unsigned short f2bf(float f) {
    unsigned u = __float_as_uint(f);
    unsigned r = u + 0x7FFFu + ((u >> 16) & 1u);   // RNE
    return (unsigned short)(r >> 16);
}
// order-preserving float <-> uint (for atomicMax-based segment max)
__device__ inline unsigned f2ord(float f) {
    unsigned u = __float_as_uint(f);
    return (u & 0x80000000u) ? ~u : (u | 0x80000000u);
}
__device__ inline float ord2f(unsigned u) {
    unsigned v = (u & 0x80000000u) ? (u & 0x7FFFFFFFu) : ~u;
    return __uint_as_float(v);
}
__device__ inline void edge_sd(const long long* __restrict__ ei, int e, int& s, int& d) {
    if (e < N_EDGES) { s = (int)ei[e]; d = (int)ei[N_EDGES + e]; }
    else             { s = d = e - N_EDGES; }      // self loop
}

// ---- CDNA5 async memory->LDS copy (ASYNCcnt-tracked, no VGPR data path) ----
__device__ inline void async_load_b128(unsigned lds_off, const void* gaddr) {
    asm volatile("global_load_async_to_lds_b128 %0, %1, off"
                 :: "v"(lds_off), "v"((unsigned long long)gaddr) : "memory");
}
__device__ inline void wait_async0() {
    asm volatile("s_wait_asynccnt 0x0" ::: "memory");
}

// ---------------- fills ----------------
__global__ void fill_f32(float* p, float v, size_t n) {
    size_t i = (size_t)blockIdx.x * blockDim.x + threadIdx.x;
    if (i < n) p[i] = v;
}
__global__ void fill_u32(unsigned* p, unsigned v, size_t n) {
    size_t i = (size_t)blockIdx.x * blockDim.x + threadIdx.x;
    if (i < n) p[i] = v;
}

// ---------------- fp32 -> bf16 cast with 2D zero padding (A matrices) ----------------
__global__ void cast_pad(const float* __restrict__ in, unsigned short* __restrict__ out,
                         int rows_in, int rows_out, int cols_in, int cols_out) {
    size_t i = (size_t)blockIdx.x * blockDim.x + threadIdx.x;
    size_t total = (size_t)rows_out * cols_out;
    if (i >= total) return;
    int c = (int)(i % cols_out);
    size_t r = i / cols_out;
    float v = (r < (size_t)rows_in && c < cols_in) ? in[r * (size_t)cols_in + c] : 0.f;
    out[i] = f2bf(v);
}

// -------- fp32 -> bf16 K-pair-packed + transposed cast (B matrices) --------
// out[c, kp] (column-panel major, [cols][K2]) = { lo = bf(in[2kp,c]), hi = bf(in[2kp+1,c]) }
__global__ void pack_pairs_t(const float* __restrict__ in, unsigned* __restrict__ out,
                             int rows_in, int K2, int cols) {
    size_t i = (size_t)blockIdx.x * blockDim.x + threadIdx.x;
    size_t total = (size_t)cols * K2;
    if (i >= total) return;
    int kp = (int)(i % K2);
    int c  = (int)(i / K2);
    int k0 = 2 * kp, k1 = k0 + 1;
    unsigned lo = (k0 < rows_in) ? f2bf(in[(size_t)k0 * cols + c]) : 0u;
    unsigned hi = (k1 < rows_in) ? f2bf(in[(size_t)k1 * cols + c]) : 0u;
    out[i] = lo | (hi << 16);
}

// ---------------- bf16 WMMA GEMM: C[M,N] = A[M,K] * B[K,N] ----------------
// A : bf16 row-major, >= gridDim.y*BM rows (pad rows zeroed) -> unconditional loads.
// Bpt: K-pair packed dwords, column-panel major [N][K/2].
// Staging: double-buffered global_load_async_to_lds_b128, compute overlaps transfers.
#define BM 256
#define BN 128
#define BK 32
#define LROW 20   // padded LDS row stride in dwords (16 data + 4 pad, 16B-aligned)

__global__ __launch_bounds__(256)
void gemm_bf16_wmma(const unsigned short* __restrict__ A,
                    const unsigned* __restrict__ Bpt,
                    float* __restrict__ C,
                    int M, int N, int K, int lda, int ldb2, int ldc) {
    __shared__ unsigned int As[2][BM * LROW];   // [buf][row][kp]  (2 x 20 KB)
    __shared__ unsigned int Bs[2][BN * LROW];   // [buf][col][kp]  (2 x 10 KB)

    const int t       = threadIdx.x;
    const int lane    = t & 31;
    const int wave    = t >> 5;
    const int waveM   = wave & 3;          // 4 waves along M, 64 rows each
    const int waveN   = wave >> 2;         // 2 waves along N, 64 cols each
    const int blockM  = blockIdx.y * BM;
    const int blockN  = blockIdx.x * BN;
    const int halfsel = lane >> 4;         // lane 0-15 vs 16-31
    const int l15     = lane & 15;
    const int lda2    = lda >> 1;          // A viewed as packed dwords along K
    const unsigned int* A2 = (const unsigned int*)A;

    v8f acc[4][4];
    #pragma unroll
    for (int i = 0; i < 4; ++i)
        #pragma unroll
        for (int j = 0; j < 4; ++j)
            #pragma unroll
            for (int r = 0; r < 8; ++r) acc[i][j][r] = 0.0f;

    // async-stage one BK tile into buffer `buf`
    auto stage = [&](int buf, int k0) {
        const int kp0 = k0 >> 1;
        // A tile: 256 rows x 16 dwords -> 1024 x 16B lane-transfers
        #pragma unroll
        for (int i = 0; i < 4; ++i) {
            int flat = t + i * 256;
            int row  = flat >> 2;
            int q    = (flat & 3) * 4;
            unsigned lds = (unsigned)(size_t)(&As[buf][row * LROW + q]);
            async_load_b128(lds, A2 + (size_t)(blockM + row) * lda2 + kp0 + q);
        }
        // B tile: 128 cols x 16 dwords -> 512 x 16B lane-transfers
        #pragma unroll
        for (int i = 0; i < 2; ++i) {
            int flat = t + i * 256;
            int col  = flat >> 2;
            int q    = (flat & 3) * 4;
            unsigned lds = (unsigned)(size_t)(&Bs[buf][col * LROW + q]);
            async_load_b128(lds, Bpt + (size_t)(blockN + col) * ldb2 + kp0 + q);
        }
    };

    stage(0, 0);
    wait_async0();
    __syncthreads();

    const int nk = K / BK;
    for (int kt = 0; kt < nk; ++kt) {
        const int buf = kt & 1;
        if (kt + 1 < nk) stage(buf ^ 1, (kt + 1) * BK);   // overlap with compute

        // ---- A fragments (ISA 16-bit A 16x32 layout): 2 x ds_load_b128 each
        Frag af[4];
        #pragma unroll
        for (int mt = 0; mt < 4; ++mt) {
            int aRow = waveM * 64 + mt * 16 + l15;
            #pragma unroll
            for (int v = 0; v < 8; ++v) {
                int kp = ((v >> 2) << 3) + (halfsel << 2) + (v & 3);
                af[mt].u[v] = As[buf][aRow * LROW + kp];
            }
        }
        // ---- B fragments (ISA 16-bit B 32x16 layout): same kp order as A -> b128 loads
        #pragma unroll
        for (int nt = 0; nt < 4; ++nt) {
            Frag bfr;
            int bCol = waveN * 64 + nt * 16 + l15;
            #pragma unroll
            for (int v = 0; v < 8; ++v)
                bfr.u[v] = Bs[buf][bCol * LROW + ((v >> 2) << 3) + (halfsel << 2) + (v & 3)];
            #pragma unroll
            for (int mt = 0; mt < 4; ++mt)
                acc[mt][nt] = __builtin_amdgcn_wmma_f32_16x16x32_bf16(
                    false, af[mt].b, false, bfr.b, (short)0, acc[mt][nt], false, false);
        }

        wait_async0();       // next tile landed in LDS
        __syncthreads();     // all waves done reading current buffer
    }

    // ---- store (ISA 32-bit C/D 16x16 layout: VGPR r -> M = r + 8*halfsel)
    #pragma unroll
    for (int mt = 0; mt < 4; ++mt)
        #pragma unroll
        for (int nt = 0; nt < 4; ++nt) {
            int col = blockN + waveN * 64 + nt * 16 + l15;
            #pragma unroll
            for (int r = 0; r < 8; ++r) {
                int row = blockM + waveM * 64 + mt * 16 + (halfsel << 3) + r;
                if (row < M && col < N)
                    C[(size_t)row * ldc + col] = acc[mt][nt][r];
            }
        }
}

// ---------------- attention coefficients: a = sum_c h[n,h,c]*att[h,c] ----------------
// grid = N_NODES, block = heads*32 (one wave per head)
__global__ void attn_coef(const float* __restrict__ h, const float* __restrict__ att_src,
                          const float* __restrict__ att_dst,
                          float* __restrict__ a_src, float* __restrict__ a_dst,
                          int heads, int C) {
    int n    = blockIdx.x;
    int head = threadIdx.x >> 5;
    int lane = threadIdx.x & 31;
    const float* hp = h + ((size_t)n * heads + head) * C;
    const float* as = att_src + (size_t)head * C;
    const float* ad = att_dst + (size_t)head * C;
    float ss = 0.f, sd = 0.f;
    for (int c = lane; c < C; c += 32) {
        float v = hp[c];
        ss += v * as[c];
        sd += v * ad[c];
    }
    #pragma unroll
    for (int m = 16; m > 0; m >>= 1) {
        ss += __shfl_xor(ss, m, 32);
        sd += __shfl_xor(sd, m, 32);
    }
    if (lane == 0) { a_src[n * heads + head] = ss; a_dst[n * heads + head] = sd; }
}

// ---------------- per-edge score e = leaky_relu(a_src[src]+a_dst[dst]) + segment max ----
__global__ void edge_score(const long long* __restrict__ ei,
                           const float* __restrict__ a_src, const float* __restrict__ a_dst,
                           float* __restrict__ e_out, unsigned* __restrict__ emax, int heads) {
    int i = blockIdx.x * blockDim.x + threadIdx.x;
    if (i >= E_TOT * heads) return;
    int e = i / heads, h = i - e * heads;
    int s, d; edge_sd(ei, e, s, d);
    float v = a_src[s * heads + h] + a_dst[d * heads + h];
    v = (v > 0.f) ? v : v * NEG_SLOPE;
    e_out[i] = v;
    atomicMax(&emax[d * heads + h], f2ord(v));     // global_atomic_max_u32, L2-resident
}

// ---------------- softmax denominator ----------------
__global__ void edge_denom(const long long* __restrict__ ei,
                           const float* __restrict__ e_in, const unsigned* __restrict__ emax,
                           float* __restrict__ denom, int heads) {
    int i = blockIdx.x * blockDim.x + threadIdx.x;
    if (i >= E_TOT * heads) return;
    int e = i / heads, h = i - e * heads;
    int s, d; edge_sd(ei, e, s, d);
    (void)s;
    float m = ord2f(emax[d * heads + h]);
    atomicAdd(&denom[d * heads + h], __expf(e_in[i] - m));
}

// ---------------- weighted scatter: out[dst] += alpha * h[src] ----------------
// grid = E_TOT, block = 256; F = heads*C. Atomic target fits in 192MB L2.
__global__ void aggregate(const long long* __restrict__ ei,
                          const float* __restrict__ h, const float* __restrict__ e_in,
                          const unsigned* __restrict__ emax, const float* __restrict__ denom,
                          float* __restrict__ out, int heads, int C) {
    int e = blockIdx.x;
    int s, d; edge_sd(ei, e, s, d);
    int F = heads * C;
    for (int j = threadIdx.x; j < F; j += blockDim.x) {
        int hh = j / C;
        float m     = ord2f(emax[d * heads + hh]);
        float alpha = __expf(e_in[e * heads + hh] - m) / (denom[d * heads + hh] + 1e-16f);
        atomicAdd(&out[(size_t)d * F + j], h[(size_t)s * F + j] * alpha);
    }
}

// ---------------- bias + relu + cast to bf16 (feeds GEMM2; pad rows -> 0) -------------
__global__ void bias_relu_cast(const float* __restrict__ in, const float* __restrict__ bias,
                               unsigned short* __restrict__ outb, int F, int rows_in,
                               size_t total) {
    size_t i = (size_t)blockIdx.x * blockDim.x + threadIdx.x;
    if (i >= total) return;
    size_t r = i / F;
    int    c = (int)(i % F);
    float v = 0.f;
    if (r < (size_t)rows_in) {
        v = in[r * F + c] + bias[c];
        v = v > 0.f ? v : 0.f;
    }
    outb[i] = f2bf(v);
}

// ---------------- global mean pool ----------------
__global__ void pool_acc(const float* __restrict__ out2, const float* __restrict__ b2,
                         const long long* __restrict__ batch,
                         float* __restrict__ gsum, float* __restrict__ gcnt) {
    int n = blockIdx.x;
    int g = (int)batch[n];
    for (int c = threadIdx.x; c < OUT_CH; c += blockDim.x)
        atomicAdd(&gsum[g * OUT_CH + c], out2[(size_t)n * OUT_CH + c] + b2[c]);
    if (threadIdx.x == 0) atomicAdd(&gcnt[g], 1.0f);
}
__global__ void pool_fin(const float* __restrict__ gsum, const float* __restrict__ gcnt,
                         float* __restrict__ out) {
    int i = blockIdx.x * blockDim.x + threadIdx.x;
    if (i >= N_GRAPHS * OUT_CH) return;
    out[i] = gsum[i] / fmaxf(gcnt[i / OUT_CH], 1.0f);
}

// =====================================================================
extern "C" void kernel_launch(void* const* d_in, const int* in_sizes, int n_in,
                              void* d_out, int out_size, void* d_ws, size_t ws_size,
                              hipStream_t stream) {
    const float*     x        = (const float*)d_in[0];
    const long long* ei       = (const long long*)d_in[1];
    const long long* batch    = (const long long*)d_in[2];
    const float*     W1       = (const float*)d_in[3];
    const float*     att_src1 = (const float*)d_in[4];
    const float*     att_dst1 = (const float*)d_in[5];
    const float*     b1       = (const float*)d_in[6];
    const float*     W2       = (const float*)d_in[7];
    const float*     att_src2 = (const float*)d_in[8];
    const float*     att_dst2 = (const float*)d_in[9];
    const float*     b2       = (const float*)d_in[10];
    float*           out      = (float*)d_out;

    // ---- carve workspace
    char* w = (char*)d_ws;
    auto take = [&](size_t bytes) { char* p = w; w += (bytes + 255) & ~(size_t)255; return p; };
    unsigned short* xb    = (unsigned short*)take((size_t)MPAD * KPAD1 * 2);
    unsigned*       w1p   = (unsigned*)take((size_t)F1 * (KPAD1 / 2) * 4);      // [N][K/2]
    float*          h1    = (float*)take((size_t)N_NODES * F1 * 4);
    float*          as1   = (float*)take((size_t)N_NODES * HEADS * 4);
    float*          ad1   = (float*)take((size_t)N_NODES * HEADS * 4);
    float*          e1    = (float*)take((size_t)E_TOT * HEADS * 4);
    unsigned*       emax1 = (unsigned*)take((size_t)N_NODES * HEADS * 4);
    float*          den1  = (float*)take((size_t)N_NODES * HEADS * 4);
    float*          out1  = (float*)take((size_t)N_NODES * F1 * 4);
    unsigned short* h1b   = (unsigned short*)take((size_t)MPAD * F1 * 2);
    unsigned*       w2p   = (unsigned*)take((size_t)OUT_CH * (F1 / 2) * 4);     // [N][K/2]
    float*          h2    = (float*)take((size_t)N_NODES * OUT_CH * 4);
    float*          as2   = (float*)take((size_t)N_NODES * 4);
    float*          ad2   = (float*)take((size_t)N_NODES * 4);
    float*          e2    = (float*)take((size_t)E_TOT * 4);
    unsigned*       emax2 = (unsigned*)take((size_t)N_NODES * 4);
    float*          den2  = (float*)take((size_t)N_NODES * 4);
    float*          out2  = (float*)take((size_t)N_NODES * OUT_CH * 4);
    float*          gsum  = (float*)take((size_t)N_GRAPHS * OUT_CH * 4);
    float*          gcnt  = (float*)take((size_t)N_GRAPHS * 4);

    auto blks = [](size_t n, int bs) { return (unsigned)((n + bs - 1) / bs); };

    // ===== layer 1 =====
    cast_pad<<<blks((size_t)MPAD * KPAD1, 256), 256, 0, stream>>>(x, xb, N_NODES, MPAD, IN_CH, KPAD1);
    pack_pairs_t<<<blks((size_t)F1 * (KPAD1 / 2), 256), 256, 0, stream>>>(W1, w1p, IN_CH, KPAD1 / 2, F1);
    {   // h1 = x @ W1  (168 GFLOP, WMMA bf16 -> f32, async-LDS double-buffered)
        dim3 g(F1 / BN, MPAD / BM);
        gemm_bf16_wmma<<<g, 256, 0, stream>>>(xb, w1p, h1, N_NODES, F1, KPAD1, KPAD1, KPAD1 / 2, F1);
    }
    attn_coef<<<N_NODES, HEADS * 32, 0, stream>>>(h1, att_src1, att_dst1, as1, ad1, HEADS, HID);
    fill_u32<<<blks((size_t)N_NODES * HEADS, 256), 256, 0, stream>>>(emax1, 0u, (size_t)N_NODES * HEADS);
    fill_f32<<<blks((size_t)N_NODES * HEADS, 256), 256, 0, stream>>>(den1, 0.f, (size_t)N_NODES * HEADS);
    edge_score<<<blks((size_t)E_TOT * HEADS, 256), 256, 0, stream>>>(ei, as1, ad1, e1, emax1, HEADS);
    edge_denom<<<blks((size_t)E_TOT * HEADS, 256), 256, 0, stream>>>(ei, e1, emax1, den1, HEADS);
    fill_f32<<<blks((size_t)N_NODES * F1, 256), 256, 0, stream>>>(out1, 0.f, (size_t)N_NODES * F1);
    aggregate<<<E_TOT, 256, 0, stream>>>(ei, h1, e1, emax1, den1, out1, HEADS, HID);
    bias_relu_cast<<<blks((size_t)MPAD * F1, 256), 256, 0, stream>>>(out1, b1, h1b, F1, N_NODES, (size_t)MPAD * F1);

    // ===== layer 2 =====
    pack_pairs_t<<<blks((size_t)OUT_CH * (F1 / 2), 256), 256, 0, stream>>>(W2, w2p, F1, F1 / 2, OUT_CH);
    {
        dim3 g(OUT_CH / BN, MPAD / BM);
        gemm_bf16_wmma<<<g, 256, 0, stream>>>(h1b, w2p, h2, N_NODES, OUT_CH, F1, F1, F1 / 2, OUT_CH);
    }
    attn_coef<<<N_NODES, 32, 0, stream>>>(h2, att_src2, att_dst2, as2, ad2, 1, OUT_CH);
    fill_u32<<<blks(N_NODES, 256), 256, 0, stream>>>(emax2, 0u, N_NODES);
    fill_f32<<<blks(N_NODES, 256), 256, 0, stream>>>(den2, 0.f, N_NODES);
    edge_score<<<blks(E_TOT, 256), 256, 0, stream>>>(ei, as2, ad2, e2, emax2, 1);
    edge_denom<<<blks(E_TOT, 256), 256, 0, stream>>>(ei, e2, emax2, den2, 1);
    fill_f32<<<blks((size_t)N_NODES * OUT_CH, 256), 256, 0, stream>>>(out2, 0.f, (size_t)N_NODES * OUT_CH);
    aggregate<<<E_TOT, 256, 0, stream>>>(ei, h2, e2, emax2, den2, out2, 1, OUT_CH);

    // ===== global mean pool =====
    fill_f32<<<blks((size_t)N_GRAPHS * OUT_CH, 256), 256, 0, stream>>>(gsum, 0.f, (size_t)N_GRAPHS * OUT_CH);
    fill_f32<<<1, 64, 0, stream>>>(gcnt, 0.f, N_GRAPHS);
    pool_acc<<<N_NODES, 256, 0, stream>>>(out2, b2, batch, gsum, gcnt);
    pool_fin<<<blks((size_t)N_GRAPHS * OUT_CH, 256), 256, 0, stream>>>(gsum, gcnt, out);
}
